// DocEncoder_1185410973873
// MI455X (gfx1250) — compile-verified, matrix operand order checked by reading, not run on hardware
//
#include <hip/hip_runtime.h>
#include <hip/hip_bf16.h>
#include <stdint.h>

// ---------------- problem constants ----------------
#define BATCH     512
#define SEQLEN    512
#define VOCAB     50265
#define PAD_TOK   1

typedef __attribute__((ext_vector_type(16))) _Float16 v16h;
typedef __attribute__((ext_vector_type(8)))  float    v8f;
typedef __attribute__((ext_vector_type(4)))  unsigned int u32x4;
typedef __attribute__((ext_vector_type(8)))  int      i32x8;
typedef __attribute__((ext_vector_type(4)))  int      i32x4;

// ---------------------------------------------------------------------------
// Kernel 0: zero the global accumulator scalar.
// ---------------------------------------------------------------------------
__global__ void k_init(float* __restrict__ total) {
  if (threadIdx.x == 0) *total = 0.0f;
}

// ---------------------------------------------------------------------------
// Kernel 1: doc_lengths total via TDM (tensor_load_to_lds) + WMMA row-sums.
// One wave (32 threads) handles 16 docs. A = (id != PAD) indicator (f16),
// B = all-ones -> D rows = doc lengths (exact integers, f32 accum).
// C/D layout: vgpr r, lanes 0-15 -> M=r ; lanes 16-31 -> M=8+r.
// A layout: lane l feeds row (l%16); since B==1, K ordering is irrelevant.
// ---------------------------------------------------------------------------
__global__ void __launch_bounds__(32)
k_doclen(const int* __restrict__ ids, float* __restrict__ dl, float* __restrict__ total) {
  __shared__ int tile[16 * SEQLEN];              // 32 KB

  const int base = blockIdx.x * 16;              // first doc of this tile

  // ---- Tensor DMA Descriptor (ISA 08_async_tensor 8.3/8.4): 2D 16x512 i32 tile
  unsigned long long ga = (unsigned long long)(uintptr_t)(const void*)(ids + (size_t)base * SEQLEN);
  unsigned int lds_lo   = (unsigned int)(uintptr_t)(void*)&tile[0]; // LDS byte offset

  u32x4 g0;
  g0[0] = 1u;                                              // count=1 (valid), user mode
  g0[1] = lds_lo;                                          // lds_addr
  g0[2] = (unsigned int)(ga & 0xFFFFFFFFull);              // global_addr[31:0]
  g0[3] = (unsigned int)((ga >> 32) & 0x01FFFFFFull)       // global_addr[56:32]
        | (2u << 30);                                      // type=2 ("image")

  i32x8 g1;
  g1[0] = (int)(2u << 16);                 // workgroup_mask=0, data_size=2 (4 bytes)
  g1[1] = (int)((unsigned)SEQLEN << 16);   // tensor_dim0 lo16 = 512
  g1[2] = (int)(16u << 16);                // tensor_dim0 hi = 0 | tensor_dim1 lo16 = 16
  g1[3] = (int)((unsigned)SEQLEN << 16);   // tensor_dim1 hi = 0 | tile_dim0 = 512
  g1[4] = 16;                              // tile_dim1 = 16, tile_dim2 = 0 (unused)
  g1[5] = SEQLEN;                          // tensor_dim0_stride lo32 = 512 elements
  g1[6] = 0;
  g1[7] = 0;

  i32x4 g2 = {0, 0, 0, 0};                 // dims 2/3 unused (tile_dim3 = 0)
  i32x4 g3 = {0, 0, 0, 0};

#if __clang_major__ >= 23
  i32x8 g4 = {0, 0, 0, 0, 0, 0, 0, 0};
  __builtin_amdgcn_tensor_load_to_lds(g0, g1, g2, g3, g4, 0);
#else
  __builtin_amdgcn_tensor_load_to_lds(g0, g1, g2, g3, 0);
#endif
  __builtin_amdgcn_s_wait_tensorcnt(0);       // DMA complete -> LDS valid
  asm volatile("" ::: "memory");              // keep ds loads below the wait

  const int lane = threadIdx.x;
  const int doc  = lane & 15;                 // row this lane feeds
  const int half = lane >> 4;                 // which 16 of the 32 K values

  v16h ones;
  #pragma unroll
  for (int j = 0; j < 16; ++j) ones[j] = (_Float16)1.0f;

  v8f acc = {};
  #pragma unroll 4
  for (int kc = 0; kc < SEQLEN; kc += 32) {
    const int* p = &tile[doc * SEQLEN + kc + half * 16];
    v16h a;
    #pragma unroll
    for (int j = 0; j < 16; ++j)
      a[j] = (p[j] != PAD_TOK) ? (_Float16)1.0f : (_Float16)0.0f;
    acc = __builtin_amdgcn_wmma_f32_16x16x32_f16(
        /*neg_a=*/false, a, /*neg_b=*/false, ones,
        /*c_mod=*/(short)0, acc, /*reuse_a=*/false, /*reuse_b=*/false);
  }

  // lanes 0 and 16 hold column N=0: vgpr r -> doc (base + half*8 + r)
  if ((lane & 15) == 0) {
    #pragma unroll
    for (int r = 0; r < 8; ++r) {
      int d = base + half * 8 + r;
      float v = acc[r];                       // exact integer <= 512
      dl[d] = v;
      atomicAdd(total, v);                    // exact -> order-independent
    }
  }
}

// ---------------------------------------------------------------------------
// Kernel 2: fused per-doc LDS histogram + BM25 transform + streamed store.
// One block per document. Counts live as packed u16 pairs in ~98 KB of LDS
// (320 KB/WGP on CDNA5 -> 3 concurrent workgroups). Counts <= 512, so a
// u32 atomic add of (1 << 16*(tok&1)) can never carry into the other half.
// Only HBM traffic: 1 MB ids read + 103 MB output write.
// ---------------------------------------------------------------------------
#define NPAIR 25136                 // dwords: 50272 u16 slots >= VOCAB (+pad)
#define NVEC  (NPAIR / 4)           // 6284 uint4 = 100,544 bytes LDS

__global__ void __launch_bounds__(512)
k_fused(const int* __restrict__ ids, float* __restrict__ out,
        const float* __restrict__ total_p) {
  __shared__ uint4 cntv[NVEC];

  const int b   = blockIdx.x;
  const int tid = threadIdx.x;

  // ---- zero the LDS histogram (b128 stores) ----
  #pragma unroll 4
  for (int i = tid; i < NVEC; i += 512) {
    uint4 z; z.x = 0u; z.y = 0u; z.z = 0u; z.w = 0u;
    cntv[i] = z;
  }
  __syncthreads();

  // ---- histogram: one token per thread ----
  unsigned* c32 = (unsigned*)cntv;
  const int tok = ids[(size_t)b * SEQLEN + tid];
  const bool nonpad = (tok != PAD_TOK);
  if (nonpad)
    atomicAdd(&c32[(unsigned)tok >> 1], 1u << (((unsigned)tok & 1u) * 16u));

  // barrier + exact per-doc length in one op
  const int dlen = __syncthreads_count(nonpad);

  // ---- BM25 per-doc constant (mirrors reference ordering) ----
  const float avgdl = (*total_p) * (1.0f / (float)BATCH);  // exact mean (pow2 div)
  const float d_avg = (float)dlen / avgdl;
  const float adder = 1.6f * (1.0f - 0.75f + 0.75f * d_avg);
  const float kp1   = 1.6f + 1.0f;                          // 2.6

  // ---- stream transformed counts straight to HBM (coalesced b32, NT) ----
  const unsigned short* c16 = (const unsigned short*)cntv;
  float* row = out + (size_t)b * VOCAB;
  for (int v = tid; v < VOCAB; v += 512) {
    float cf  = (float)c16[v];
    float val = cf * kp1 / (cf + adder);      // 0 count -> exact 0 (matches ref)
    __builtin_nontemporal_store(val, &row[v]);
  }
}

// ---------------------------------------------------------------------------
extern "C" void kernel_launch(void* const* d_in, const int* in_sizes, int n_in,
                              void* d_out, int out_size, void* d_ws, size_t ws_size,
                              hipStream_t stream) {
  (void)in_sizes; (void)n_in; (void)out_size; (void)ws_size;
  const int* ids = (const int*)d_in[0];     // [512, 512] int32
  // d_in[1] = beta: unused by the reference computation.

  float* out   = (float*)d_out;             // [512, 50265] f32
  float* dl    = (float*)d_ws;              // 512 doc lengths (debug/aux)
  float* total = dl + BATCH;                // 1 scalar: sum of doc lengths

  k_init  <<<1,          32, 0, stream>>>(total);
  k_doclen<<<BATCH / 16, 32, 0, stream>>>(ids, dl, total);
  k_fused <<<BATCH,     512, 0, stream>>>(ids, out, total);
}
